// GlobalPointer_15238543966612
// MI455X (gfx1250) — compile-verified, matrix operand order checked by reading, not run on hardware
//
#include <hip/hip_runtime.h>
#include <stdint.h>

typedef __bf16 bf16;
typedef bf16 bf16x8  __attribute__((ext_vector_type(8)));
typedef bf16 bf16x16 __attribute__((ext_vector_type(16)));
typedef float v8f    __attribute__((ext_vector_type(8)));

union Frag { bf16x8 h[2]; bf16x16 v; };

#define WMMA_BF16(A, B, C) \
  __builtin_amdgcn_wmma_f32_16x16x32_bf16(false, (A), false, (B), (short)0, (C), false, false)

// Async global->LDS copy of 16B (gfx1250, ASYNCcnt-tracked).
// lds_off: wave-relative LDS byte offset (= low 32 bits of generic shared ptr).
__device__ __forceinline__ void async_copy_b128(uint32_t lds_off, const void* gptr) {
  asm volatile("global_load_async_to_lds_b128 %0, %1, off"
               :: "v"(lds_off), "v"((uint64_t)(uintptr_t)gptr) : "memory");
}
__device__ __forceinline__ void async_wait0() {
  asm volatile("s_wait_asynccnt 0x0" ::: "memory");
}

// ---------------------------------------------------------------------------
// Kernel 1: x = inputs @ W + b, split into q/k heads, apply RoPE, emit
// Q/K as (hi,lo) bf16 planes laid out [b*12+h][pos][d].
// inputs: (8192 x 768) fp32,  W: (768 x 1536) fp32.
// Block tile: 128(M) x 64(N), K-chunks of 32 (one WMMA K per chunk).
// Staging needs a VALU pass (fp32 -> bf16 hi/lo split), so it uses normal
// loads + ds stores with global_prefetch of the next K-chunk.
// ---------------------------------------------------------------------------
__global__ __launch_bounds__(256)
void proj_rope_kernel(const float* __restrict__ inp, const float* __restrict__ Wg,
                      const float* __restrict__ bias,
                      bf16* __restrict__ qhi, bf16* __restrict__ qlo,
                      bf16* __restrict__ khi, bf16* __restrict__ klo)
{
  // padded row strides (bf16 units): 40*2B = 80B, multiple of 16B for b128 reads
  __shared__ __align__(16) bf16 Ah[128][40];
  __shared__ __align__(16) bf16 Al[128][40];
  __shared__ __align__(16) bf16 Bh[64][40];   // W^T tile: [n][k]
  __shared__ __align__(16) bf16 Bl[64][40];

  const int tid  = threadIdx.x;
  const int lane = tid & 31;
  const int wid  = tid >> 5;
  const int m0   = blockIdx.y * 128;
  const int n0   = blockIdx.x * 64;

  const v8f zf = {0.f,0.f,0.f,0.f,0.f,0.f,0.f,0.f};
  v8f acc[4];
#pragma unroll
  for (int s = 0; s < 4; ++s) acc[s] = zf;

  const int lm   = lane & 15;
  const int half = lane >> 4;
  const int mrow = (wid << 4) + lm;
  const int ksel = half ? 8 : 0;        // ISA 16-bit A layout: lanes>=16 hold K 8..15 / 24..31

  for (int k0 = 0; k0 < 768; k0 += 32) {
    // prefetch next K-chunk of A and W into caches (global_prefetch_b8)
    if (k0 + 32 < 768) {
      __builtin_prefetch(&inp[(size_t)(m0 + (tid >> 1)) * 768 + (k0 + 32) + ((tid & 1) << 4)], 0, 1);
      __builtin_prefetch(&Wg[(size_t)(k0 + 32 + (tid >> 3)) * 1536 + n0 + ((tid & 7) << 3)], 0, 1);
    }
    // stage A (128x32) as hi/lo bf16
#pragma unroll
    for (int p = 0; p < 16; ++p) {
      int idx = tid + p * 256;          // 0..4095
      int r = idx >> 5, c = idx & 31;
      float v = inp[(size_t)(m0 + r) * 768 + k0 + c];
      bf16 h = (bf16)v;
      Ah[r][c] = h;
      Al[r][c] = (bf16)(v - (float)h);
    }
    // stage W^T (64n x 32k) as hi/lo bf16 (coalesced along n)
#pragma unroll
    for (int p = 0; p < 8; ++p) {
      int idx = tid + p * 256;          // 0..2047
      int j = idx & 63, c = idx >> 6;
      float v = Wg[(size_t)(k0 + c) * 1536 + n0 + j];
      bf16 h = (bf16)v;
      Bh[j][c] = h;
      Bl[j][c] = (bf16)(v - (float)h);
    }
    __syncthreads();

    Frag ah, al;
    ah.h[0] = *(const bf16x8*)&Ah[mrow][ksel];
    ah.h[1] = *(const bf16x8*)&Ah[mrow][ksel + 16];
    al.h[0] = *(const bf16x8*)&Al[mrow][ksel];
    al.h[1] = *(const bf16x8*)&Al[mrow][ksel + 16];
#pragma unroll
    for (int s = 0; s < 4; ++s) {
      int col = (s << 4) + lm;
      Frag bh2, bl2;
      bh2.h[0] = *(const bf16x8*)&Bh[col][ksel];
      bh2.h[1] = *(const bf16x8*)&Bh[col][ksel + 16];
      bl2.h[0] = *(const bf16x8*)&Bl[col][ksel];
      bl2.h[1] = *(const bf16x8*)&Bl[col][ksel + 16];
      // Ootomo split: hi*lo + lo*hi + hi*hi, fp32 accumulate
      acc[s] = WMMA_BF16(al.v, bh2.v, acc[s]);
      acc[s] = WMMA_BF16(ah.v, bl2.v, acc[s]);
      acc[s] = WMMA_BF16(ah.v, bh2.v, acc[s]);
    }
    __syncthreads();
  }

  // Epilogue: bias + RoPE + hi/lo split store.
  // C layout: VGPR i -> M = i + 8*(lane>=16), N = lane&15.
#pragma unroll
  for (int s = 0; s < 4; ++s) {
    int col = n0 + (s << 4) + lm;       // 0..1535
    float bcol = bias[col];
    int d   = col & 63;
    int j2  = d & ~1;
    // 10000^(-j2/64) = exp(-ln(10000)/64 * j2)
    float inv = __expf(-0.14391156831212787f * (float)j2);
    int head  = col >> 7;               // uniform across lanes (16-col tile)
    int is_k  = (col >> 6) & 1;         // uniform across lanes
    bf16* __restrict__ hp = is_k ? khi : qhi;
    bf16* __restrict__ lp = is_k ? klo : qlo;
#pragma unroll
    for (int i = 0; i < 8; ++i) {
      int m    = m0 + (wid << 4) + i + (half << 3);
      int bidx = m >> 10;
      int pos  = m & 1023;
      float x  = acc[s][i] + bcol;
      float prt = __shfl_xor(x, 1, 32);     // partner column (2j <-> 2j+1), same row
      float ang = (float)pos * inv;
      float sn, cs; __sincosf(ang, &sn, &cs);
      float rot = (d & 1) ? prt : -prt;     // rotate_half
      float y = x * cs + rot * sn;
      bf16 h = (bf16)y;
      bf16 l = (bf16)(y - (float)h);
      size_t oidx = ((size_t)((bidx * 12 + head) * 1024 + pos)) * 64 + d;
      hp[oidx] = h;
      lp[oidx] = l;
    }
  }
}

// ---------------------------------------------------------------------------
// Kernel 2: logits[b,h,m,n] = (Q[m]·K[n]) with mask, causal(-1e12, m>n), /8.
// Block tile: 128(M) x 64(N) per (b,h); K = 64 = two WMMA chunks.
// Tile staging is a pure byte copy of pre-split bf16 planes -> use gfx1250
// GLOBAL_LOAD_ASYNC_TO_LDS_B128 (ASYNCcnt), no VGPR round-trip.
// ---------------------------------------------------------------------------
__global__ __launch_bounds__(256)
void attn_logits_kernel(const bf16* __restrict__ qhi, const bf16* __restrict__ qlo,
                        const bf16* __restrict__ khi, const bf16* __restrict__ klo,
                        const float* __restrict__ mask, float* __restrict__ out)
{
  __shared__ __align__(16) bf16 Qh[128][72];   // 72*2B = 144B stride, 16B aligned
  __shared__ __align__(16) bf16 Ql[128][72];
  __shared__ __align__(16) bf16 Kh[64][72];
  __shared__ __align__(16) bf16 Kl[64][72];
  __shared__ float Mrow[128];
  __shared__ float Mcol[64];

  const int tid  = threadIdx.x;
  const int lane = tid & 31;
  const int wid  = tid >> 5;
  const int bh   = blockIdx.z;          // b*12 + h
  const int b    = bh / 12;
  const int m0   = blockIdx.y * 128;
  const int n0   = blockIdx.x * 64;

  // Stage Q (128x64) and K (64x64) hi/lo tiles as async 16B chunks.
  const bf16* qh_src = qhi + ((size_t)bh * 1024 + m0) * 64;
  const bf16* ql_src = qlo + ((size_t)bh * 1024 + m0) * 64;
  const bf16* kh_src = khi + ((size_t)bh * 1024 + n0) * 64;
  const bf16* kl_src = klo + ((size_t)bh * 1024 + n0) * 64;
#pragma unroll
  for (int p = 0; p < 4; ++p) {
    int idx = tid + p * 256;            // 0..1023 chunks of 8 bf16
    int r = idx >> 3, c = (idx & 7) << 3;
    async_copy_b128((uint32_t)(uintptr_t)&Qh[r][c], &qh_src[r * 64 + c]);
    async_copy_b128((uint32_t)(uintptr_t)&Ql[r][c], &ql_src[r * 64 + c]);
  }
#pragma unroll
  for (int p = 0; p < 2; ++p) {
    int idx = tid + p * 256;            // 0..511 chunks
    int r = idx >> 3, c = (idx & 7) << 3;
    async_copy_b128((uint32_t)(uintptr_t)&Kh[r][c], &kh_src[r * 64 + c]);
    async_copy_b128((uint32_t)(uintptr_t)&Kl[r][c], &kl_src[r * 64 + c]);
  }
  if (tid < 128) Mrow[tid] = mask[(size_t)b * 1024 + m0 + tid];
  if (tid < 64)  Mcol[tid] = mask[(size_t)b * 1024 + n0 + tid];
  async_wait0();                        // ASYNCcnt == 0 for this wave
  __syncthreads();                      // make all waves' tiles visible

  const v8f zf = {0.f,0.f,0.f,0.f,0.f,0.f,0.f,0.f};
  v8f acc[4];
#pragma unroll
  for (int s = 0; s < 4; ++s) acc[s] = zf;

  const int lm   = lane & 15;
  const int half = lane >> 4;
  const int mrow = (wid << 4) + lm;
#pragma unroll
  for (int kk = 0; kk < 64; kk += 32) {
    const int e0 = kk + (half ? 8 : 0);
    Frag ah, al;
    ah.h[0] = *(const bf16x8*)&Qh[mrow][e0];
    ah.h[1] = *(const bf16x8*)&Qh[mrow][e0 + 16];
    al.h[0] = *(const bf16x8*)&Ql[mrow][e0];
    al.h[1] = *(const bf16x8*)&Ql[mrow][e0 + 16];
#pragma unroll
    for (int s = 0; s < 4; ++s) {
      int col = (s << 4) + lm;
      Frag bh2, bl2;
      bh2.h[0] = *(const bf16x8*)&Kh[col][e0];
      bh2.h[1] = *(const bf16x8*)&Kh[col][e0 + 16];
      bl2.h[0] = *(const bf16x8*)&Kl[col][e0];
      bl2.h[1] = *(const bf16x8*)&Kl[col][e0 + 16];
      acc[s] = WMMA_BF16(al.v, bh2.v, acc[s]);
      acc[s] = WMMA_BF16(ah.v, bl2.v, acc[s]);
      acc[s] = WMMA_BF16(ah.v, bh2.v, acc[s]);
    }
  }

  // Epilogue: mask, causal, scale (exactly in reference order).
#pragma unroll
  for (int s = 0; s < 4; ++s) {
    int nloc = (s << 4) + lm;
    int n = n0 + nloc;
    float m2 = Mcol[nloc];
#pragma unroll
    for (int i = 0; i < 8; ++i) {
      int mloc = (wid << 4) + i + (half << 3);
      int m = m0 + mloc;
      float m1 = Mrow[mloc];
      float x = acc[s][i];
      x = x * m1 - 1.0e12f * (1.0f - m1);
      x = x * m2 - 1.0e12f * (1.0f - m2);
      if (m > n) x -= 1.0e12f;
      out[((size_t)bh * 1024 + m) * 1024 + n] = x * 0.125f;
    }
  }
}

// ---------------------------------------------------------------------------
extern "C" void kernel_launch(void* const* d_in, const int* in_sizes, int n_in,
                              void* d_out, int out_size, void* d_ws, size_t ws_size,
                              hipStream_t stream) {
  const float* inp  = (const float*)d_in[0];   // (8,1024,768) fp32
  const float* mask = (const float*)d_in[1];   // (8,1024)     fp32
  const float* Wg   = (const float*)d_in[2];   // (768,1536)   fp32
  const float* bias = (const float*)d_in[3];   // (1536,)      fp32
  float* out = (float*)d_out;                  // (8,12,1024,1024) fp32

  const size_t PLANE = (size_t)96 * 1024 * 64; // elements per bf16 plane
  bf16* qhi = (bf16*)d_ws;
  bf16* qlo = qhi + PLANE;
  bf16* khi = qlo + PLANE;
  bf16* klo = khi + PLANE;

  // GEMM1 + RoPE: grid = (1536/64 n-tiles, 8192/128 m-tiles)
  proj_rope_kernel<<<dim3(24, 64), 256, 0, stream>>>(inp, Wg, bias, qhi, qlo, khi, klo);
  // QK^T + mask/causal/scale: grid = (1024/64 n-tiles, 1024/128 m-tiles, 96 bh)
  attn_logits_kernel<<<dim3(16, 8, 96), 256, 0, stream>>>(qhi, qlo, khi, klo, mask, out);
}
